// UnmaskedHead_35081292873919
// MI455X (gfx1250) — matrix operand order repaired and checked by reading, not compiled, not run on hardware
//
#include <hip/hip_runtime.h>
#include <math.h>

// MI455X / gfx1250 fused attention:
//   kernel 1: QKV projection (f32 -> bf16, WMMA f32_16x16x32_bf16)
//   kernel 2: flash attention, double-buffered K/V tiles in LDS,
//             async global->LDS copies for K (ASYNCcnt) when available.
// Workspace: 3 * B*T*H bf16 = 6 MB.

typedef __attribute__((ext_vector_type(16))) __bf16 v16bf;
typedef __attribute__((ext_vector_type(8)))  float  v8f;
typedef __attribute__((ext_vector_type(4)))  int    v4i;

#define B_SZ  4
#define T_LEN 4096
#define C_DIM 1024
#define H_DIM 64
#define BR    64   // query rows per block (flash)
#define BC    64   // key rows per iteration (flash)

#if defined(__gfx1250__) && __has_builtin(__builtin_amdgcn_global_load_async_to_lds_b128)
#define USE_ASYNC_LDS 1
#else
#define USE_ASYNC_LDS 0
#endif

typedef __attribute__((address_space(1))) v4i gv4i_t;   // global 4-dword
typedef __attribute__((address_space(3))) v4i sv4i_t;   // LDS 4-dword

static __device__ __forceinline__ void wait_async0() {
#if USE_ASYNC_LDS
#if __has_builtin(__builtin_amdgcn_s_wait_asynccnt)
  __builtin_amdgcn_s_wait_asynccnt(0);
#else
  asm volatile("s_wait_asynccnt 0x0" ::: "memory");
#endif
#endif
}

static __device__ __forceinline__ v8f wmma_bf16(v16bf a, v16bf b, v8f c) {
  return __builtin_amdgcn_wmma_f32_16x16x32_bf16(
      /*neg_a=*/false, a, /*neg_b=*/false, b,
      /*c_mod=*/(short)0, c, /*reuse_a=*/false, /*reuse_b=*/false);
}

// A-matrix fragment 16x32 bf16 from row-major storage (ISA 7.12.2):
// lane(0-15): rows 0-15, K in {hi*8 .. hi*8+7} then {16+hi*8 ..}, hi = lane>>4
static __device__ __forceinline__ v16bf load_A_lds(const __bf16* src, int stride) {
  int lane = threadIdx.x & 31;
  int row  = lane & 15;
  int hi   = lane >> 4;
  const __bf16* r = src + row * stride + hi * 8;
  v16bf a;
#pragma unroll
  for (int i = 0; i < 8; ++i) a[i] = r[i];
#pragma unroll
  for (int i = 0; i < 8; ++i) a[8 + i] = r[16 + i];
  return a;
}

// B-matrix fragment 32x16 bf16 where logical B[k][n] = srcT[n*stride + k]
// (transposed storage; K in {hi*8 .. hi*8+7} then {16+hi*8 ..}, matching A).
static __device__ __forceinline__ v16bf load_B_lds(const __bf16* srcT, int stride) {
  int lane = threadIdx.x & 31;
  int n    = lane & 15;
  int hi   = lane >> 4;
  const __bf16* r = srcT + n * stride + hi * 8;
  v16bf b;
#pragma unroll
  for (int i = 0; i < 8; ++i) b[i] = r[i];
#pragma unroll
  for (int i = 0; i < 8; ++i) b[8 + i] = r[16 + i];
  return b;
}

static __device__ __forceinline__ float half16_max(float v) {
#pragma unroll
  for (int m = 8; m >= 1; m >>= 1) v = fmaxf(v, __shfl_xor(v, m, 32));
  return v;
}
static __device__ __forceinline__ float half16_sum(float v) {
#pragma unroll
  for (int m = 8; m >= 1; m >>= 1) v += __shfl_xor(v, m, 32);
  return v;
}

// ---------------------------------------------------------------------------
// Kernel 1: y = x @ W for W in {Wq, Wk, Wv} (blockIdx.y selects), bf16 output.
// ---------------------------------------------------------------------------
__global__ __launch_bounds__(256)
void qkv_proj_kernel(const float* __restrict__ x,
                     const float* __restrict__ Wq,
                     const float* __restrict__ Wk,
                     const float* __restrict__ Wv,
                     __bf16* __restrict__ Qo,
                     __bf16* __restrict__ Ko,
                     __bf16* __restrict__ Vo) {
  __shared__ __bf16 xb[128][32];   // x tile, row-major
  __shared__ __bf16 wt[64][32];    // W chunk, TRANSPOSED: wt[n][k]

  const float* W   = (blockIdx.y == 0) ? Wq : (blockIdx.y == 1) ? Wk : Wv;
  __bf16*      out = (blockIdx.y == 0) ? Qo : (blockIdx.y == 1) ? Ko : Vo;

  const int tid  = threadIdx.x;
  const int lane = tid & 31;
  const int wave = tid >> 5;
  const int row0 = blockIdx.x * 128;

  v8f acc[4];
#pragma unroll
  for (int nt = 0; nt < 4; ++nt)
#pragma unroll
    for (int r = 0; r < 8; ++r) acc[nt][r] = 0.0f;

  for (int k0 = 0; k0 < C_DIM; k0 += 32) {
    __syncthreads();
    // x tile: 128x32 f32 = 1024 float4; 4 per thread
#pragma unroll
    for (int it = 0; it < 4; ++it) {
      int idx = tid + it * 256;
      int r   = idx >> 3;
      int c4  = idx & 7;
      float4 d = *(const float4*)(x + (size_t)(row0 + r) * C_DIM + k0 + c4 * 4);
      xb[r][c4 * 4 + 0] = (__bf16)d.x;
      xb[r][c4 * 4 + 1] = (__bf16)d.y;
      xb[r][c4 * 4 + 2] = (__bf16)d.z;
      xb[r][c4 * 4 + 3] = (__bf16)d.w;
    }
    // W chunk: 32x64 f32 = 512 float4; 2 per thread; store transposed
#pragma unroll
    for (int it = 0; it < 2; ++it) {
      int idx = tid + it * 256;
      int r   = idx >> 4;
      int c4  = idx & 15;
      float4 d = *(const float4*)(W + (size_t)(k0 + r) * H_DIM + c4 * 4);
      wt[c4 * 4 + 0][r] = (__bf16)d.x;
      wt[c4 * 4 + 1][r] = (__bf16)d.y;
      wt[c4 * 4 + 2][r] = (__bf16)d.z;
      wt[c4 * 4 + 3][r] = (__bf16)d.w;
    }
    __syncthreads();

    v16bf a = load_A_lds(&xb[wave * 16][0], 32);
#pragma unroll
    for (int nt = 0; nt < 4; ++nt) {
      v16bf b = load_B_lds(&wt[nt * 16][0], 32);
      acc[nt] = wmma_bf16(a, b, acc[nt]);
    }
  }

  const int hi = lane >> 4;
  const int n  = lane & 15;
#pragma unroll
  for (int nt = 0; nt < 4; ++nt)
#pragma unroll
    for (int r = 0; r < 8; ++r) {
      int m = wave * 16 + r + 8 * hi;
      out[(size_t)(row0 + m) * H_DIM + nt * 16 + n] = (__bf16)acc[nt][r];
    }
}

// ---------------------------------------------------------------------------
// Kernel 2: flash attention, double-buffered LDS tiles.
// Block = 128 threads (4 waves), each wave owns 16 query rows.
// ---------------------------------------------------------------------------
__global__ __launch_bounds__(128)
void flash_attn_kernel(const __bf16* __restrict__ Q,
                       const __bf16* __restrict__ K,
                       const __bf16* __restrict__ V,
                       float* __restrict__ out,
                       float scale) {
  __shared__ __bf16 Klds[2][BC][H_DIM];   // row-major: [key][feat]
  __shared__ __bf16 Vt[2][H_DIM][BC];     // transposed: [feat][key]
  __shared__ __bf16 Plds[4][16][BC];      // per-wave P buffer

  const int tid   = threadIdx.x;
  const int lane  = tid & 31;
  const int wave  = tid >> 5;
  const int hi    = lane >> 4;
  const int nlo   = lane & 15;
  const int batch = blockIdx.y;
  const int qrow0 = blockIdx.x * BR + wave * 16;

  const __bf16* Qb = Q + (size_t)batch * T_LEN * H_DIM;
  const __bf16* Kb = K + (size_t)batch * T_LEN * H_DIM;
  const __bf16* Vb = V + (size_t)batch * T_LEN * H_DIM;

  // tile-copy coordinates: thread covers rows rbase+16p, 16-byte column c8*16
  const int rbase = tid >> 3;
  const int c8    = tid & 7;

  union HU { uint4 u; __bf16 h[8]; };
  uint4 vreg[4];
#if !USE_ASYNC_LDS
  uint4 kreg[4];
#endif

  auto issue_k = [&](int jb, int buf) {
#pragma unroll
    for (int p = 0; p < 4; ++p) {
      int r = rbase + p * 16;
      const __bf16* g = Kb + (size_t)(jb * BC + r) * H_DIM + c8 * 8;
#if USE_ASYNC_LDS
      __builtin_amdgcn_global_load_async_to_lds_b128(
          (gv4i_t*)(void*)g, (sv4i_t*)(void*)&Klds[buf][r][c8 * 8], 0, 0);
#else
      kreg[p] = *(const uint4*)g;
#endif
    }
#if !USE_ASYNC_LDS
    (void)buf;
#endif
  };
  auto load_v = [&](int jb) {
#pragma unroll
    for (int p = 0; p < 4; ++p)
      vreg[p] = *(const uint4*)(Vb + (size_t)(jb * BC + rbase + p * 16) * H_DIM + c8 * 8);
  };
  auto store_tiles = [&](int buf) {
#pragma unroll
    for (int p = 0; p < 4; ++p) {
      int r = rbase + p * 16;
      HU t; t.u = vreg[p];
#pragma unroll
      for (int c = 0; c < 8; ++c) Vt[buf][c8 * 8 + c][r] = t.h[c];  // transpose
#if !USE_ASYNC_LDS
      *(uint4*)(&Klds[buf][r][c8 * 8]) = kreg[p];
#endif
    }
  };

  // Q fragments (16 rows x 64 feats = two 16x32 A-frags), registers whole kernel
  v16bf qfrag[2];
  {
    const __bf16* src = Qb + (size_t)(qrow0 + nlo) * H_DIM;
#pragma unroll
    for (int kc = 0; kc < 2; ++kc) {
      const __bf16* r = src + kc * 32 + hi * 8;
#pragma unroll
      for (int i = 0; i < 8; ++i) qfrag[kc][i] = r[i];
#pragma unroll
      for (int i = 0; i < 8; ++i) qfrag[kc][8 + i] = r[16 + i];
    }
  }

  v8f   o[4];
  float mrun[8], lrun[8];
#pragma unroll
  for (int nt = 0; nt < 4; ++nt)
#pragma unroll
    for (int r = 0; r < 8; ++r) o[nt][r] = 0.0f;
#pragma unroll
  for (int r = 0; r < 8; ++r) { mrun[r] = -1e30f; lrun[r] = 0.0f; }

  // ---- prologue: tile 0 into buffer 0 ----
  const int NB = T_LEN / BC;
  issue_k(0, 0);
  load_v(0);
  store_tiles(0);
  wait_async0();
  __syncthreads();

  for (int jb = 0; jb < NB; ++jb) {
    const int  cur  = jb & 1;
    const bool more = (jb + 1 < NB);

    // issue next tile's global traffic before compute (latency overlap)
    if (more) { issue_k(jb + 1, cur ^ 1); load_v(jb + 1); }

    // ---- S = (Q Kt) * scale : four 16x16 tiles per wave ----
    v8f s[4];
#pragma unroll
    for (int nt = 0; nt < 4; ++nt) {
#pragma unroll
      for (int r = 0; r < 8; ++r) s[nt][r] = 0.0f;
#pragma unroll
      for (int kc = 0; kc < 2; ++kc) {
        v16bf b = load_B_lds(&Klds[cur][nt * 16][kc * 32], H_DIM);
        s[nt] = wmma_bf16(qfrag[kc], b, s[nt]);
      }
#pragma unroll
      for (int r = 0; r < 8; ++r) s[nt][r] *= scale;
    }

    // ---- online softmax: row m = r + 8*hi lives in 16 lanes sharing hi ----
#pragma unroll
    for (int r = 0; r < 8; ++r) {
      float tmax = fmaxf(fmaxf(s[0][r], s[1][r]), fmaxf(s[2][r], s[3][r]));
      tmax = half16_max(tmax);
      float newm = fmaxf(mrun[r], tmax);
      float corr = __expf(mrun[r] - newm);
      float psum = 0.0f;
#pragma unroll
      for (int nt = 0; nt < 4; ++nt) {
        float p = __expf(s[nt][r] - newm);
        s[nt][r] = p;
        psum += p;
      }
      psum = half16_sum(psum);
      lrun[r] = lrun[r] * corr + psum;
      mrun[r] = newm;
#pragma unroll
      for (int nt = 0; nt < 4; ++nt) o[nt][r] *= corr;
    }

    // ---- C/D-layout -> A-layout via per-wave LDS buffer ----
#pragma unroll
    for (int nt = 0; nt < 4; ++nt)
#pragma unroll
      for (int r = 0; r < 8; ++r)
        Plds[wave][r + 8 * hi][nt * 16 + nlo] = (__bf16)s[nt][r];
    asm volatile("s_wait_dscnt 0" ::: "memory");   // wave-local LDS RAW

    v16bf pa0 = load_A_lds(&Plds[wave][0][0],  BC);
    v16bf pa1 = load_A_lds(&Plds[wave][0][32], BC);
#pragma unroll
    for (int nt = 0; nt < 4; ++nt) {
      o[nt] = wmma_bf16(pa0, load_B_lds(&Vt[cur][nt * 16][0],  BC), o[nt]);
      o[nt] = wmma_bf16(pa1, load_B_lds(&Vt[cur][nt * 16][32], BC), o[nt]);
    }

    // ---- commit next tile into the other buffer ----
    if (more) store_tiles(cur ^ 1);
    wait_async0();
    __syncthreads();
  }

  // ---- epilogue: normalize and store f32 ----
#pragma unroll
  for (int nt = 0; nt < 4; ++nt)
#pragma unroll
    for (int r = 0; r < 8; ++r) {
      int m = r + 8 * hi;
      out[((size_t)batch * T_LEN + qrow0 + m) * H_DIM + nt * 16 + nlo] =
          o[nt][r] / lrun[r];
    }
}

// ---------------------------------------------------------------------------
extern "C" void kernel_launch(void* const* d_in, const int* in_sizes, int n_in,
                              void* d_out, int out_size, void* d_ws, size_t ws_size,
                              hipStream_t stream) {
  const float* x  = (const float*)d_in[0];
  const float* Wk = (const float*)d_in[1];
  const float* Wq = (const float*)d_in[2];
  const float* Wv = (const float*)d_in[3];
  float* out = (float*)d_out;

  const size_t elems = (size_t)B_SZ * T_LEN * H_DIM;   // 1,048,576
  __bf16* Qb = (__bf16*)d_ws;
  __bf16* Kb = Qb + elems;
  __bf16* Vb = Kb + elems;                              // 6 MB total bf16

  dim3 g1((B_SZ * T_LEN) / 128, 3);
  qkv_proj_kernel<<<g1, 256, 0, stream>>>(x, Wq, Wk, Wv, Qb, Kb, Vb);

  const float scale = 1.0f / sqrtf((float)C_DIM);       // C^-0.5 = 1/32
  dim3 g2(T_LEN / BR, B_SZ);
  flash_attn_kernel<<<g2, 128, 0, stream>>>(Qb, Kb, Vb, out, scale);
}
